// POMv2_10771777978558
// MI455X (gfx1250) — matrix operand order repaired and checked by reading, not compiled, not run on hardware
//
#include <hip/hip_runtime.h>
#include <hip/hip_bf16.h>

// ---------------- problem constants ----------------
#define NN   60000      // nodes
#define NE   120000     // edges
#define NG   2048       // graphs
#define HD   192        // hidden dim
#define NLAY 5
#define NDESC 200

typedef __attribute__((ext_vector_type(16))) _Float16 v16h;
typedef __attribute__((ext_vector_type(8)))  float    v8f;

// ---- optional CDNA5 async global->LDS path (guarded; falls back to manual copy) ----
#if __has_builtin(__builtin_amdgcn_global_load_async_to_lds_b128) && \
    __has_builtin(__builtin_amdgcn_s_wait_asynccnt)
#define USE_ASYNC_LDS 1
#else
#define USE_ASYNC_LDS 0
#endif

#if USE_ASYNC_LDS
typedef int v4i_t __attribute__((vector_size(16)));
typedef __attribute__((address_space(1))) v4i_t as1_v4i;
typedef __attribute__((address_space(3))) v4i_t as3_v4i;
__device__ __forceinline__ void async_copy16(const void* g, void* l) {
    __builtin_amdgcn_global_load_async_to_lds_b128((as1_v4i*)g, (as3_v4i*)l, 0, 0);
}
__device__ __forceinline__ void async_wait0() {
    __builtin_amdgcn_s_wait_asynccnt(0);
}
#endif

__device__ __forceinline__ float gelu_f(float v) {
    return 0.5f * v * (1.0f + erff(v * 0.70710678118654752f));
}

__device__ __forceinline__ v8f wmma_f16(v16h a, v16h b, v8f c) {
    return __builtin_amdgcn_wmma_f32_16x16x32_f16(false, a, false, b,
                                                  (short)0, c, false, false);
}

// A fragment 16x32 f16 from row-major LDS [m][k] (ld halves).
// lanes 0-15: M=lane, halves = K{k0..k0+15}
// lanes 16-31: M=lane-16, halves = K{k0+16..k0+31}
__device__ __forceinline__ v16h frag_a_lds(const _Float16* A, int m0, int k0, int ld) {
    const int lane = threadIdx.x & 31;
    const _Float16* p = A + (m0 + (lane & 15)) * ld + k0 + ((lane >> 4) << 4);
    union { uint4 u[2]; v16h v; } t;
    t.u[0] = *(const uint4*)p;
    t.u[1] = *(const uint4*)(p + 8);
    return t.v;
}

// B fragment 32x16 (KxN) f16 from [n][k]-transposed matrix (global or LDS).
// lanes 0-15: N=lane, K=k0..k0+15 ; lanes 16-31: N=lane-16, K=k0+16..k0+31
__device__ __forceinline__ v16h frag_b(const _Float16* BT, int n0, int k0, int ld) {
    const int lane = threadIdx.x & 31;
    const _Float16* p = BT + (size_t)(n0 + (lane & 15)) * ld + k0 + ((lane >> 4) << 4);
    union { uint4 u[2]; v16h v; } t;
    t.u[0] = *(const uint4*)p;
    t.u[1] = *(const uint4*)(p + 8);
    return t.v;
}

__device__ __forceinline__ void zero8(v8f& v) {
#pragma unroll
    for (int i = 0; i < 8; ++i) v[i] = 0.0f;
}

// ---------------- block LayerNorm helpers (192 threads / block) ----------------
__device__ __forceinline__ float block_sum_192(float v, float* red) {
    const int t = threadIdx.x;
    __syncthreads();
    red[t] = v;
    __syncthreads();
    for (int s = 96; s >= 3; s >>= 1) {
        if (t < s) red[t] += red[t + s];
        __syncthreads();
    }
    return red[0] + red[1] + red[2];
}

__device__ __forceinline__ float ln_192(float v, float* red, float g, float b) {
    const float mu  = block_sum_192(v, red) * (1.0f / 192.0f);
    const float d   = v - mu;
    const float var = block_sum_192(d * d, red) * (1.0f / 192.0f);
    return d * rsqrtf(var + 1e-5f) * g + b;
}

// ---------------- one-time weight convert+transpose to f16 [n][k] ----------------
__global__ void prep_w_kernel(const float* __restrict__ w1, const float* __restrict__ w2,
                              const float* __restrict__ wu,
                              _Float16* __restrict__ w1T, _Float16* __restrict__ w2T,
                              _Float16* __restrict__ wuT) {
    const long S1 = 384L * 384, S2 = 192L * 384;
    long i = (long)blockIdx.x * blockDim.x + threadIdx.x;
    if (i < NLAY * S1) {
        const long l = i / S1, r = i % S1, n = r / 384, k = r % 384;
        w1T[i] = (_Float16)w1[l * 385 * 384 + k * 384 + n];
        return;
    }
    i -= NLAY * S1;
    if (i < NLAY * S2) {
        const long l = i / S2, r = i % S2, n = r / 384, k = r % 384;
        w2T[i] = (_Float16)w2[l * 384 * 192 + k * 192 + n];
        return;
    }
    i -= NLAY * S2;
    const long l = i / S2, r = i % S2, n = r / 384, k = r % 384;
    wuT[i] = (_Float16)wu[l * 384 * 192 + k * 192 + n];
}

// ---------------- init: zero agg/pools, set maxpool to enc(-inf) ----------------
__global__ void init_kernel(float* agg, float* psum, unsigned* pmax, float* pcnt) {
    const long i = (long)blockIdx.x * blockDim.x + threadIdx.x;
    if (i < (long)NN * HD) agg[i] = 0.0f;
    if (i < (long)NG * HD) { psum[i] = 0.0f; pmax[i] = 0x007FFFFFu; } // enc(-inf)
    if (i < NG) pcnt[i] = 0.0f;
}

// ---------------- encoder: x = LN(GELU(af @ enc_w + b)) ----------------
__global__ void encode_kernel(const float* __restrict__ af,
                              const float* __restrict__ w, const float* __restrict__ b,
                              const float* __restrict__ g, const float* __restrict__ bt,
                              float* __restrict__ x, _Float16* __restrict__ xh) {
    __shared__ float red[HD];
    const int node = blockIdx.x, t = threadIdx.x;
    const float a0 = af[node * 4 + 0], a1 = af[node * 4 + 1];
    const float a2 = af[node * 4 + 2], a3 = af[node * 4 + 3];
    float v = a0 * w[t] + a1 * w[HD + t] + a2 * w[2 * HD + t] + a3 * w[3 * HD + t] + b[t];
    v = gelu_f(v);
    const float o = ln_192(v, red, g[t], bt[t]);
    x[node * HD + t]  = o;
    xh[node * HD + t] = (_Float16)o;
}

// ---------------- fused message MLP + scatter (per layer) ----------------
// hidden = GELU([x_src|x_dst] @ W1 + e*w1row + b1) ; m = hidden @ W2 + b2
// agg[dst] += m   (32 edges per block, 8 waves; weights read as f16 [n][k] from L2)
__global__ __launch_bounds__(256) void msg_kernel(
    const _Float16* __restrict__ xh, const int* __restrict__ eidx,
    const float* __restrict__ eattr,
    const _Float16* __restrict__ w1T, const float* __restrict__ w1row,
    const float* __restrict__ b1,
    const _Float16* __restrict__ w2T, const float* __restrict__ b2,
    float* __restrict__ agg) {
    __shared__ _Float16 Ae[32 * 384];    // gathered [x_src | x_dst], f16
    __shared__ _Float16 Hid[32 * 384];   // gelu(hidden), f16
    __shared__ float    sAttr[32];
    __shared__ int      sDst[32];

    const int tid = threadIdx.x, wave = tid >> 5, lane = tid & 31;
    const int e0 = blockIdx.x * 32;
    const int mt = wave >> 2, wq = wave & 3;

    if (tid < 32) {
        sAttr[tid] = eattr[e0 + tid];
        sDst[tid]  = eidx[NE + e0 + tid];
    }
    // gather endpoint rows as 16B chunks: 32 edges x (24 src + 24 dst)
#if USE_ASYNC_LDS
    for (int idx = tid; idx < 32 * 48; idx += 256) {
        const int ei = idx / 48, c = idx % 48;
        const int node = (c < 24) ? eidx[e0 + ei] : eidx[NE + e0 + ei];
        const int cc = (c < 24) ? c : (c - 24);
        async_copy16(xh + (size_t)node * HD + cc * 8,
                     Ae + ei * 384 + ((c < 24) ? 0 : 192) + cc * 8);
    }
#else
    for (int idx = tid; idx < 32 * 48; idx += 256) {
        const int ei = idx / 48, c = idx % 48;
        const int node = (c < 24) ? eidx[e0 + ei] : eidx[NE + e0 + ei];
        const int cc = (c < 24) ? c : (c - 24);
        *(uint4*)(Ae + ei * 384 + ((c < 24) ? 0 : 192) + cc * 8) =
            *(const uint4*)(xh + (size_t)node * HD + cc * 8);
    }
#endif
    // warm WGP$ with the gemm2 weight slab while gemm1 runs
    __builtin_prefetch(w2T + (size_t)tid * 288, 0, 1);
#if USE_ASYNC_LDS
    async_wait0();
#endif
    __syncthreads();

    // ---- GEMM1: [32x384] x [384x384]; B frags straight from global (L2-resident) ----
    v8f acc[6];
#pragma unroll
    for (int j = 0; j < 6; ++j) zero8(acc[j]);

    for (int ks = 0; ks < 12; ++ks) {
        const v16h a = frag_a_lds(Ae, mt * 16, ks * 32, 384);
#pragma unroll
        for (int j = 0; j < 6; ++j) {
            const v16h b = frag_b(w1T, (wq * 6 + j) * 16, ks * 32, 384);
            acc[j] = wmma_f16(a, b, acc[j]);
        }
    }

    // bias + edge rank-1 term + GELU -> Hid (f16)
    {
        const int ncol = lane & 15, moff = (lane >> 4) << 3;
#pragma unroll
        for (int j = 0; j < 6; ++j) {
            const int n = (wq * 6 + j) * 16 + ncol;
            const float bn = b1[n], we = w1row[n];
#pragma unroll
            for (int r = 0; r < 8; ++r) {
                const int m = mt * 16 + moff + r;
                Hid[m * 384 + n] = (_Float16)gelu_f(acc[j][r] + bn + sAttr[m] * we);
            }
        }
    }
    __syncthreads();

    // ---- GEMM2: [32x384] x [384x192] ----
    v8f acc2[3];
#pragma unroll
    for (int j = 0; j < 3; ++j) zero8(acc2[j]);

    for (int ks = 0; ks < 12; ++ks) {
        const v16h a = frag_a_lds(Hid, mt * 16, ks * 32, 384);
#pragma unroll
        for (int j = 0; j < 3; ++j) {
            const v16h b = frag_b(w2T, (wq * 3 + j) * 16, ks * 32, 384);
            acc2[j] = wmma_f16(a, b, acc2[j]);
        }
    }

    // scatter-add per-edge message to agg[dst]
    {
        const int ncol = lane & 15, moff = (lane >> 4) << 3;
#pragma unroll
        for (int j = 0; j < 3; ++j) {
            const int n = (wq * 3 + j) * 16 + ncol;
            const float bn = b2[n];
#pragma unroll
            for (int r = 0; r < 8; ++r) {
                const int m = mt * 16 + moff + r;
                atomicAdd(&agg[(size_t)sDst[m] * HD + n], acc2[j][r] + bn);
            }
        }
    }
}

// ---------------- update GEMM: y = [x|agg] @ upd_w + b ----------------
__global__ __launch_bounds__(256) void upd_gemm_kernel(
    const _Float16* __restrict__ xh, const float* __restrict__ agg,
    const _Float16* __restrict__ wT, const float* __restrict__ b,
    float* __restrict__ y) {
    __shared__ _Float16 An[32 * 384];
    const int tid = threadIdx.x, wave = tid >> 5, lane = tid & 31;
    const int r0 = blockIdx.x * 32;
    const int mt = wave >> 2, wq = wave & 3;

    // stage xh half (pure copy)
#if USE_ASYNC_LDS
    for (int idx = tid; idx < 32 * 24; idx += 256) {
        const int i = idx / 24, c = idx % 24;
        async_copy16(xh + (size_t)(r0 + i) * HD + c * 8, An + i * 384 + c * 8);
    }
#else
    for (int idx = tid; idx < 32 * 24; idx += 256) {
        const int i = idx / 24, c = idx % 24;
        *(uint4*)(An + i * 384 + c * 8) =
            *(const uint4*)(xh + (size_t)(r0 + i) * HD + c * 8);
    }
#endif
    // stage agg half (fp32 -> f16)
    for (int idx = tid; idx < 32 * 48; idx += 256) {
        const int i = idx / 48, c = idx % 48;
        const float4 f = ((const float4*)(agg + (size_t)(r0 + i) * HD))[c];
        union { _Float16 h[4]; uint2 u; } pk;
        pk.h[0] = (_Float16)f.x; pk.h[1] = (_Float16)f.y;
        pk.h[2] = (_Float16)f.z; pk.h[3] = (_Float16)f.w;
        *(uint2*)(An + i * 384 + 192 + c * 4) = pk.u;
    }
#if USE_ASYNC_LDS
    async_wait0();
#endif
    __syncthreads();

    v8f acc[3];
#pragma unroll
    for (int j = 0; j < 3; ++j) zero8(acc[j]);

    for (int ks = 0; ks < 12; ++ks) {
        const v16h a = frag_a_lds(An, mt * 16, ks * 32, 384);
#pragma unroll
        for (int j = 0; j < 3; ++j) {
            const v16h b = frag_b(wT, (wq * 3 + j) * 16, ks * 32, 384);
            acc[j] = wmma_f16(a, b, acc[j]);
        }
    }

    const int ncol = lane & 15, moff = (lane >> 4) << 3;
#pragma unroll
    for (int j = 0; j < 3; ++j) {
        const int n = (wq * 3 + j) * 16 + ncol;
        const float bn = b[n];
#pragma unroll
        for (int r = 0; r < 8; ++r) {
            const int m = r0 + mt * 16 + moff + r;
            y[(size_t)m * HD + n] = acc[j][r] + bn;
        }
    }
}

// ---------------- pointwise update: x = LN(x + GELU(y)); zero agg ----------------
__global__ void upd_point_kernel(const float* __restrict__ y,
                                 float* __restrict__ x, _Float16* __restrict__ xh,
                                 const float* __restrict__ g, const float* __restrict__ bt,
                                 float* __restrict__ agg) {
    __shared__ float red[HD];
    const int node = blockIdx.x, t = threadIdx.x;
    const float v = x[node * HD + t] + gelu_f(y[node * HD + t]);
    const float o = ln_192(v, red, g[t], bt[t]);
    x[node * HD + t]  = o;
    xh[node * HD + t] = (_Float16)o;
    agg[node * HD + t] = 0.0f;
}

// ---------------- pooling ----------------
__global__ void pool_kernel(const float* __restrict__ x, const int* __restrict__ batch,
                            float* __restrict__ psum, unsigned* __restrict__ pmax,
                            float* __restrict__ pcnt) {
    const int node = blockIdx.x, t = threadIdx.x;
    const int gid = batch[node];
    const float v = x[node * HD + t];
    atomicAdd(&psum[gid * HD + t], v);
    unsigned u = __float_as_uint(v);
    u = (u & 0x80000000u) ? ~u : (u | 0x80000000u);   // order-preserving encode
    atomicMax(&pmax[gid * HD + t], u);
    if (t == 0) atomicAdd(&pcnt[gid], 1.0f);
}

__global__ void pool_fin_kernel(const float* __restrict__ psum,
                                const unsigned* __restrict__ pmax,
                                const float* __restrict__ pcnt,
                                float* __restrict__ gf) {
    const int gid = blockIdx.x, t = threadIdx.x;
    const float c = pcnt[gid];
    const float mean = (c > 0.0f) ? psum[gid * HD + t] / c : 0.0f;
    const unsigned u = pmax[gid * HD + t];
    const float mx = (u & 0x80000000u) ? __uint_as_float(u & 0x7FFFFFFFu)
                                       : __uint_as_float(~u);
    gf[gid * 2 * HD + t]      = mean;
    gf[gid * 2 * HD + HD + t] = mx;
}

// ---------------- small head GEMMs ----------------
__global__ void head_kernel(const float* __restrict__ in, const float* __restrict__ w,
                            const float* __restrict__ b, float* __restrict__ out,
                            int K, int N, int doGelu) {
    const int gid = blockIdx.x, t = threadIdx.x;
    float acc = b[t];
    for (int k = 0; k < K; ++k) acc += in[gid * K + k] * w[k * N + t];
    out[gid * N + t] = doGelu ? gelu_f(acc) : acc;
}

// ---------------- launch ----------------
extern "C" void kernel_launch(void* const* d_in, const int* in_sizes, int n_in,
                              void* d_out, int out_size, void* d_ws, size_t ws_size,
                              hipStream_t stream) {
    const float* af       = (const float*)d_in[0];
    const float* eattr    = (const float*)d_in[1];
    const int*   eidx     = (const int*)d_in[2];
    const int*   batch    = (const int*)d_in[3];
    const float* enc_w    = (const float*)d_in[4];
    const float* enc_b    = (const float*)d_in[5];
    const float* enc_ln_g = (const float*)d_in[6];
    const float* enc_ln_b = (const float*)d_in[7];
    const float* msg_w1   = (const float*)d_in[8];
    const float* msg_b1   = (const float*)d_in[9];
    const float* msg_w2   = (const float*)d_in[10];
    const float* msg_b2   = (const float*)d_in[11];
    const float* upd_w    = (const float*)d_in[12];
    const float* upd_b    = (const float*)d_in[13];
    const float* ln_g     = (const float*)d_in[14];
    const float* ln_b     = (const float*)d_in[15];
    const float* h1_w     = (const float*)d_in[16];
    const float* h1_b     = (const float*)d_in[17];
    const float* h2_w     = (const float*)d_in[18];
    const float* h2_b     = (const float*)d_in[19];
    const float* h3_w     = (const float*)d_in[20];
    const float* h3_b     = (const float*)d_in[21];

    char* p = (char*)d_ws;
    auto carve = [&](size_t bytes) -> void* {
        void* r = (void*)p;
        p += (bytes + 255) & ~(size_t)255;
        return r;
    };
    float*     x    = (float*)    carve((size_t)NN * HD * 4);
    _Float16*  xh   = (_Float16*) carve((size_t)NN * HD * 2);
    float*     agg  = (float*)    carve((size_t)NN * HD * 4);
    float*     y    = (float*)    carve((size_t)NN * HD * 4);
    _Float16*  w1T  = (_Float16*) carve((size_t)NLAY * 384 * 384 * 2);
    _Float16*  w2T  = (_Float16*) carve((size_t)NLAY * 192 * 384 * 2);
    _Float16*  wuT  = (_Float16*) carve((size_t)NLAY * 192 * 384 * 2);
    float*     psum = (float*)    carve((size_t)NG * HD * 4);
    unsigned*  pmax = (unsigned*) carve((size_t)NG * HD * 4);
    float*     pcnt = (float*)    carve((size_t)NG * 4);
    float*     gf   = (float*)    carve((size_t)NG * 2 * HD * 4);
    float*     h1o  = (float*)    carve((size_t)NG * HD * 4);
    float*     h2o  = (float*)    carve((size_t)NG * (HD / 2) * 4);
    float*     outp = (float*)d_out;

    const long prepN = (long)NLAY * (384L * 384 + 2 * 192L * 384);
    prep_w_kernel<<<(prepN + 255) / 256, 256, 0, stream>>>(msg_w1, msg_w2, upd_w,
                                                           w1T, w2T, wuT);
    const long initN = (long)NN * HD;
    init_kernel<<<(initN + 255) / 256, 256, 0, stream>>>(agg, psum, pmax, pcnt);
    encode_kernel<<<NN, HD, 0, stream>>>(af, enc_w, enc_b, enc_ln_g, enc_ln_b, x, xh);

    for (int l = 0; l < NLAY; ++l) {
        msg_kernel<<<NE / 32, 256, 0, stream>>>(
            xh, eidx, eattr,
            w1T + (size_t)l * 384 * 384,
            msg_w1 + (size_t)l * 385 * 384 + 384 * 384,   // row 384 (edge term)
            msg_b1 + (size_t)l * 384,
            w2T + (size_t)l * 192 * 384,
            msg_b2 + (size_t)l * 192, agg);
        upd_gemm_kernel<<<NN / 32, 256, 0, stream>>>(
            xh, agg, wuT + (size_t)l * 192 * 384, upd_b + (size_t)l * 192, y);
        upd_point_kernel<<<NN, HD, 0, stream>>>(
            y, x, xh, ln_g + (size_t)l * 192, ln_b + (size_t)l * 192, agg);
    }

    pool_kernel<<<NN, HD, 0, stream>>>(x, batch, psum, pmax, pcnt);
    pool_fin_kernel<<<NG, HD, 0, stream>>>(psum, pmax, pcnt, gf);

    head_kernel<<<NG, HD,      0, stream>>>(gf,  h1_w, h1_b, h1o,  2 * HD, HD,     1);
    head_kernel<<<NG, HD / 2,  0, stream>>>(h1o, h2_w, h2_b, h2o,  HD,     HD / 2, 1);
    head_kernel<<<NG, NDESC,   0, stream>>>(h2o, h3_w, h3_b, outp, HD / 2, NDESC,  0);
}
